// Experts_5952824673136
// MI455X (gfx1250) — matrix operand order repaired and checked by reading, not compiled, not run on hardware
//
#include <hip/hip_runtime.h>
#include <math.h>

// ---------------------------------------------------------------------------
// Grouped expert FFN for MI455X (gfx1250, wave32, WMMA).
//   hidden = gelu(x @ w1 + b1)   -> f16 in d_ws
//   out    = hidden @ w2 + b2    -> f32 in d_out
// E=16, N=2048 tokens/expert, D=1024, H=4096.
// Compute-bound (~550 GFLOP vs ~35us of HBM traffic at 23.3 TB/s), so run the
// math on v_wmma_f32_16x16x32_f16 with f32 accumulation (f16 > bf16 here:
// same WMMA throughput, 8x better mantissa; all operands are O(1)).
// Register-staged software pipeline overlaps global loads with WMMA.
// ---------------------------------------------------------------------------

typedef _Float16 f16;
typedef f16   v16h __attribute__((ext_vector_type(16)));
typedef f16   v8h  __attribute__((ext_vector_type(8)));
typedef f16   v4h  __attribute__((ext_vector_type(4)));
typedef f16   v2h  __attribute__((ext_vector_type(2)));
typedef float v8f  __attribute__((ext_vector_type(8)));
typedef float v4f  __attribute__((ext_vector_type(4)));

#define BM 128
#define BN 128
#define BK 32
#define LDSS 40   // padded LDS row stride in f16 elems (80B -> 20 banks, conflict-free)

// Branchless tanh-form GELU: native v_exp_f32 + v_rcp_f32, no EXEC divergence.
__device__ __forceinline__ float gelu_fast(float x) {
    float u  = x * (1.0f + 0.044715f * x * x);          // x + 0.044715 x^3
    float y  = 0.7978845608028654f * u;                 // sqrt(2/pi) * u
    float ay = fabsf(y);
    float z  = __builtin_amdgcn_exp2f(-2.885390081777927f * ay);  // e^(-2|y|)
    float t  = (1.0f - z) * __builtin_amdgcn_rcpf(1.0f + z);      // tanh(|y|)
    t = copysignf(t, y);
    return 0.5f * x * (1.0f + t);
}

// One tiled GEMM: C[M,Nd] = A[M,K] * B[K,Nd] (+bias, optional GELU).
// A_IS_F16: A is f16 (GEMM2 reading the hidden buffer), else fp32.
// GELU_F16: apply GELU and store f16 (GEMM1), else store f32.
template <bool A_IS_F16, bool GELU_F16>
__global__ __launch_bounds__(256)
void ffn_wmma_gemm(const void* __restrict__ Aptr,
                   const float* __restrict__ Bptr,
                   const float* __restrict__ bias,
                   void* __restrict__ Cptr,
                   int M, int Nd, int K)
{
    __shared__ f16 sA[BM * LDSS];   // [row m][k]   (k-major, padded)
    __shared__ f16 sB[BN * LDSS];   // [col n][k]   (transposed, padded)

    const int e    = blockIdx.z;
    const int m0   = blockIdx.y * BM;
    const int n0   = blockIdx.x * BN;
    const int t    = threadIdx.x;
    const int lane = t & 31;
    const int wave = t >> 5;
    const int wm   = wave >> 2;      // 0..1 : 64-row slab
    const int wn   = wave & 3;       // 0..3 : 32-col slab
    const int lm   = lane & 15;
    const bool hiH = lane >= 16;

    const size_t eA = (size_t)e * M * K;
    const size_t eB = (size_t)e * K * Nd;
    const size_t eC = (size_t)e * M * Nd;

    const float* Af = (const float*)Aptr + eA;
    const f16*   Ah = (const f16*)Aptr + eA;
    const float* Bg = Bptr + eB;

    v8f acc[4][2] = {};

    // staging maps
    const int ar  = t >> 3;          // A: row 0..31 (x4 iters)
    const int ag  = t & 7;           // A: 4-elem group along K
    const int bn4 = (t & 31) * 4;    // B: 4-col group along N
    const int bkp = t >> 5;          // B: k-pair 0..7 (x2 iters)

    // -------- software-pipeline register stage --------
    v4f aRegF[4];
    v4h aRegH[4];
    v4f bReg[2][2];

    auto load_stage = [&](int k0) {
#pragma unroll
        for (int i = 0; i < 4; ++i) {
            int r = ar + i * 32;
            if (A_IS_F16)
                aRegH[i] = *(const v4h*)&Ah[(size_t)(m0 + r) * K + k0 + ag * 4];
            else
                aRegF[i] = *(const v4f*)&Af[(size_t)(m0 + r) * K + k0 + ag * 4];
        }
#pragma unroll
        for (int i = 0; i < 2; ++i) {
            int kp = bkp + i * 8;
            const float* row0 = &Bg[(size_t)(k0 + 2 * kp) * Nd + n0 + bn4];
            bReg[i][0] = *(const v4f*)row0;
            bReg[i][1] = *(const v4f*)(row0 + Nd);
        }
    };

    auto store_stage = [&]() {
#pragma unroll
        for (int i = 0; i < 4; ++i) {
            int r = ar + i * 32;
            if (A_IS_F16) {
                *(v4h*)&sA[r * LDSS + ag * 4] = aRegH[i];
            } else {
                v4f a = aRegF[i];
                v4h p;
                p[0] = (f16)a[0]; p[1] = (f16)a[1];
                p[2] = (f16)a[2]; p[3] = (f16)a[3];
                *(v4h*)&sA[r * LDSS + ag * 4] = p;
            }
        }
#pragma unroll
        for (int i = 0; i < 2; ++i) {
            int kp = bkp + i * 8;                 // k-pair 0..15
#pragma unroll
            for (int c = 0; c < 4; ++c) {
                v2h p;
                p[0] = (f16)bReg[i][0][c];
                p[1] = (f16)bReg[i][1][c];
                *(v2h*)&sB[(bn4 + c) * LDSS + 2 * kp] = p;
            }
        }
    };

    load_stage(0);   // prologue

    for (int k0 = 0; k0 < K; k0 += BK) {
        store_stage();
        __syncthreads();

        // issue next tile's global loads; their wait lands before the next
        // store_stage, overlapping load latency with the WMMAs below
        if (k0 + BK < K) load_stage(k0 + BK);

        // ---- fragments (contiguous k = 16*(lane/16) + s for A and B) + 8x WMMA ----
        const int kbA = hiH ? 16 : 0;   // A: lanes 16-31 hold K 16..31
        const int kbB = hiH ? 16 : 0;   // B: lanes 16-31 hold K 16..31

        v16h afrag[4];
#pragma unroll
        for (int mi = 0; mi < 4; ++mi) {
            int row = wm * 64 + mi * 16 + lm;
            v8h lo  = *(const v8h*)&sA[row * LDSS + kbA];
            v8h hi8 = *(const v8h*)&sA[row * LDSS + kbA + 8];
            afrag[mi] = __builtin_shufflevector(lo, hi8,
                0,1,2,3,4,5,6,7,8,9,10,11,12,13,14,15);
        }
        v16h bfrag[2];
#pragma unroll
        for (int nj = 0; nj < 2; ++nj) {
            int col = wn * 32 + nj * 16 + lm;
            v8h lo  = *(const v8h*)&sB[col * LDSS + kbB];
            v8h hi8 = *(const v8h*)&sB[col * LDSS + kbB + 8];
            bfrag[nj] = __builtin_shufflevector(lo, hi8,
                0,1,2,3,4,5,6,7,8,9,10,11,12,13,14,15);
        }

#pragma unroll
        for (int mi = 0; mi < 4; ++mi)
#pragma unroll
            for (int nj = 0; nj < 2; ++nj)
                acc[mi][nj] = __builtin_amdgcn_wmma_f32_16x16x32_f16(
                    /*neg_a=*/false, afrag[mi],
                    /*neg_b=*/false, bfrag[nj],
                    /*c_mod=*/(short)0, acc[mi][nj],
                    /*reuse_a=*/false, /*reuse_b=*/false);

        __syncthreads();
    }

    // ---- epilogue: bias (+ GELU); C layout: vgpr r -> row r+(hi?8:0), col lane%16
#pragma unroll
    for (int nj = 0; nj < 2; ++nj) {
        int ncol = n0 + wn * 32 + nj * 16 + lm;
        float bv = bias[ncol];
#pragma unroll
        for (int mi = 0; mi < 4; ++mi) {
#pragma unroll
            for (int r = 0; r < 8; ++r) {
                int mrow = m0 + wm * 64 + mi * 16 + r + (hiH ? 8 : 0);
                float v = acc[mi][nj][r] + bv;
                if (GELU_F16) {
                    v = gelu_fast(v);
                    ((f16*)Cptr)[eC + (size_t)mrow * Nd + ncol] = (f16)v;
                } else {
                    ((float*)Cptr)[eC + (size_t)mrow * Nd + ncol] = v;
                }
            }
        }
    }
}

extern "C" void kernel_launch(void* const* d_in, const int* in_sizes, int n_in,
                              void* d_out, int out_size, void* d_ws, size_t ws_size,
                              hipStream_t stream)
{
    constexpr int E = 16, N = 2048, D = 1024, H = 4096;
    const float* x  = (const float*)d_in[0];
    const float* w1 = (const float*)d_in[1];
    const float* w2 = (const float*)d_in[2];
    const float* b1 = (const float*)d_in[3];
    const float* b2 = (const float*)d_in[4];

    f16* hidden = (f16*)d_ws;   // E*N*H f16 = 256 MiB scratch

    dim3 blk(256);
    // GEMM1 + bias + GELU -> f16 hidden
    ffn_wmma_gemm<false, true><<<dim3(H / BN, N / BM, E), blk, 0, stream>>>(
        x, w1, b1, hidden, N, H, D);
    // GEMM2 + bias -> f32 out
    ffn_wmma_gemm<true, false><<<dim3(D / BN, N / BM, E), blk, 0, stream>>>(
        hidden, w2, b2, d_out, N, D, H);
}
